// GAT_38044820308171
// MI455X (gfx1250) — matrix operand order repaired and checked by reading, not compiled, not run on hardware
//
#include <hip/hip_runtime.h>

#define NN 50000
#define EE 800000
#define HH 3
#define FDIM 128
#define HID3 384

typedef __attribute__((ext_vector_type(16))) __bf16 v16bf;
typedef __attribute__((ext_vector_type(8)))  float  v8f;

// ---------- helpers ----------
__device__ __forceinline__ __bf16 f2bf(float f) {
    unsigned u = __float_as_uint(f);
    u += 0x7FFFu + ((u >> 16) & 1u);           // round-to-nearest-even
    unsigned short h = (unsigned short)(u >> 16);
    union { unsigned short s; __bf16 b; } cv; cv.s = h; return cv.b;
}
__device__ __forceinline__ unsigned fenc(float f) {   // order-preserving float->uint
    unsigned u = __float_as_uint(f);
    return (u & 0x80000000u) ? ~u : (u | 0x80000000u);
}
__device__ __forceinline__ float fdec(unsigned u) {
    unsigned v = (u & 0x80000000u) ? (u & 0x7FFFFFFFu) : ~u;
    return __uint_as_float(v);
}

// ---------- fills ----------
__global__ void fill_f32(float* p, float v, int n) {
    int i = blockIdx.x * 256 + threadIdx.x; if (i < n) p[i] = v;
}
__global__ void fill_u32(unsigned* p, unsigned v, int n) {
    int i = blockIdx.x * 256 + threadIdx.x; if (i < n) p[i] = v;
}

// ---------- fp32 -> bf16 bulk convert (4 elems/thread) ----------
__global__ void cvt_bf16(const float* __restrict__ in, __bf16* __restrict__ out, int n4) {
    int i = blockIdx.x * 256 + threadIdx.x; if (i >= n4) return;
    float4 v = ((const float4*)in)[i];
    union { __bf16 b[4]; uint2 u; } o;
    o.b[0] = f2bf(v.x); o.b[1] = f2bf(v.y); o.b[2] = f2bf(v.z); o.b[3] = f2bf(v.w);
    ((uint2*)out)[i] = o.u;
}

// ---------- weight transpose (+zero pad rows), fp32 -> bf16 : W (K x M) -> WT (Mp x K) ----------
__global__ void transpose_pad_bf16(const float* __restrict__ W, __bf16* __restrict__ WT,
                                   int K, int M, int Mp) {
    int idx = blockIdx.x * 256 + threadIdx.x;
    if (idx >= Mp * K) return;
    int m = idx / K, k = idx - m * K;
    WT[idx] = f2bf((m < M) ? W[k * M + m] : 0.f);
}

// ---------- WMMA GEMM: out(N x Mstore) = A(N x 128, bf16) * WT^T(bf16), fused bias+leaky ----
// One wave per 16x(16*FG) output tile: 1 A fragment reused across FG B fragments.
// K = 128 fixed -> 4*FG v_wmma_f32_16x16x32_bf16 per wave.
template<int FG>
__global__ __launch_bounds__(256) void gemm_wmma(
        const __bf16* __restrict__ A, const __bf16* __restrict__ WT,
        const float* __restrict__ bias, float* __restrict__ outF, __bf16* __restrict__ outB,
        int nodeTiles, int featGroups, int ldc, int Mstore, float slope, int applyAct) {
    int wid  = (blockIdx.x * 256 + threadIdx.x) >> 5;
    if (wid >= nodeTiles * featGroups) return;           // wave-uniform: EXEC stays full
    int lane = threadIdx.x & 31;
    int tg = wid % featGroups;
    int tn = wid / featGroups;

    int m   = (lane & 15);
    int kcA = (lane >> 4) * 16;        // A frag: one 16-elem K chunk per lane
    int kcB = (lane >> 4) * 16;        // B frag: one 16-elem K chunk per lane
    const __bf16* arow = A  + (size_t)(tn * 16 + m) * FDIM;
    const __bf16* brow = WT + (size_t)(tg * FG * 16 + m) * FDIM;

    v8f acc[FG] = {};
#pragma unroll
    for (int k0 = 0; k0 < 128; k0 += 32) {
        union { uint4 q[2]; v16bf v; } ua;
        ua.q[0] = *(const uint4*)(arow + k0 + kcA);
        ua.q[1] = *(const uint4*)(arow + k0 + kcA + 8);
#pragma unroll
        for (int g = 0; g < FG; ++g) {
            union { uint4 q[2]; v16bf v; } ub;
            const __bf16* bp = brow + (size_t)g * 16 * FDIM + k0 + kcB;
            ub.q[0] = *(const uint4*)(bp);
            ub.q[1] = *(const uint4*)(bp + 8);
            acc[g] = __builtin_amdgcn_wmma_f32_16x16x32_bf16(
                         false, ua.v, false, ub.v, (short)0, acc[g], false, false);
        }
    }

    int nodeBase = tn * 16 + (lane >> 4) * 8;
#pragma unroll
    for (int g = 0; g < FG; ++g) {
        int feat   = (tg * FG + g) * 16 + (lane & 15);
        bool colOk = feat < Mstore;
        float bv = (bias != nullptr && colOk) ? bias[feat] : 0.f;
#pragma unroll
        for (int i = 0; i < 8; ++i) {
            float v = acc[g][i] + bv;
            if (applyAct) v = v > 0.f ? v : slope * v;
            if (colOk) {
                if (outB) outB[(size_t)(nodeBase + i) * ldc + feat] = f2bf(v);
                else      outF[(size_t)(nodeBase + i) * ldc + feat] = v;
            }
        }
    }
}

// ---------- el/er = einsum('nhf,hf->nh'): one wave per (n,h) ----------
__global__ __launch_bounds__(256) void attn_dots(
        const float* __restrict__ h, const float* __restrict__ al,
        const float* __restrict__ ar, float* __restrict__ el, float* __restrict__ er) {
    int wid  = (blockIdx.x * 256 + threadIdx.x) >> 5;
    int lane = threadIdx.x & 31;
    if (wid >= NN * HH) return;
    int n  = wid / HH;
    int hh = wid - n * HH;
    const float4 x = *(const float4*)(h  + (size_t)n * HID3 + hh * FDIM + lane * 4);
    const float4 a = *(const float4*)(al + hh * FDIM + lane * 4);
    const float4 r = *(const float4*)(ar + hh * FDIM + lane * 4);
    float sl = x.x*a.x + x.y*a.y + x.z*a.z + x.w*a.w;
    float sr = x.x*r.x + x.y*r.y + x.z*r.z + x.w*r.w;
#pragma unroll
    for (int off = 16; off > 0; off >>= 1) {
        sl += __shfl_xor(sl, off, 32);
        sr += __shfl_xor(sr, off, 32);
    }
    if (lane == 0) { el[wid] = sl; er[wid] = sr; }
}

// ---------- edge pass 1: score + segment max ----------
__global__ void edge_scores(const int* __restrict__ src, const int* __restrict__ dst,
                            const float* __restrict__ el, const float* __restrict__ er,
                            float* __restrict__ esc, unsigned* __restrict__ menc) {
    int e = blockIdx.x * 256 + threadIdx.x; if (e >= EE) return;
    int s = src[e], d = dst[e];
#pragma unroll
    for (int h = 0; h < HH; ++h) {
        float v = el[s * HH + h] + er[d * HH + h];
        v = v > 0.f ? v : 0.2f * v;                     // GAT leaky slope
        esc[e * HH + h] = v;
        atomicMax(&menc[d * HH + h], fenc(v));
    }
}

// ---------- edge pass 2: exp(e - m[dst]) + segment sum (esc rewritten in place) ----------
__global__ void edge_expsum(const int* __restrict__ dst, float* __restrict__ esc,
                            const unsigned* __restrict__ menc, float* __restrict__ denom) {
    int e = blockIdx.x * 256 + threadIdx.x; if (e >= EE) return;
    int d = dst[e];
#pragma unroll
    for (int h = 0; h < HH; ++h) {
        float m = fdec(menc[d * HH + h]);
        float x = __expf(esc[e * HH + h] - m);
        esc[e * HH + h] = x;
        atomicAdd(&denom[d * HH + h], x);
    }
}

// ---------- edge pass 3: agg[dst] += alpha * h[src]; thread = (e, h, 4 feats) ----------
__global__ void edge_agg(const int* __restrict__ src, const int* __restrict__ dst,
                         const float* __restrict__ hfeat, const float* __restrict__ esc,
                         const float* __restrict__ denom, float* __restrict__ agg) {
    int idx = blockIdx.x * 256 + threadIdx.x;
    if (idx >= EE * HH * 32) return;
    int e = idx / 96;
    int r = idx - e * 96;
    int h = r >> 5;
    int q = r & 31;
    int s = src[e], d = dst[e];
    float alpha = esc[e * HH + h] / (denom[d * HH + h] + 1e-9f);
    const float4 hv = *(const float4*)(hfeat + (size_t)s * HID3 + h * FDIM + (q << 2));
    float* o = agg + (size_t)d * HID3 + h * FDIM + (q << 2);
    atomicAdd(o + 0, alpha * hv.x);
    atomicAdd(o + 1, alpha * hv.y);
    atomicAdd(o + 2, alpha * hv.z);
    atomicAdd(o + 3, alpha * hv.w);
}

// ---------- (agg + b) -> leaky(0.01) -> mean over heads -> bf16 (next GEMM input) ----------
__global__ void bias_act_mean(const float* __restrict__ agg, const float* __restrict__ b,
                              __bf16* __restrict__ x) {
    int idx = blockIdx.x * 256 + threadIdx.x;
    if (idx >= NN * FDIM) return;
    int n = idx / FDIM, f = idx - n * FDIM;
    float acc = 0.f;
#pragma unroll
    for (int h = 0; h < HH; ++h) {
        float v = agg[(size_t)n * HID3 + h * FDIM + f] + b[h * FDIM + f];
        v = v > 0.f ? v : 0.01f * v;
        acc += v;
    }
    x[idx] = f2bf(acc * (1.f / 3.f));
}

// ===================== host orchestration =====================
extern "C" void kernel_launch(void* const* d_in, const int* in_sizes, int n_in,
                              void* d_out, int out_size, void* d_ws, size_t ws_size,
                              hipStream_t stream) {
    const float* in_feat = (const float*)d_in[0];
    const int*   src     = (const int*)d_in[1];
    const int*   dst     = (const int*)d_in[2];
    const float* W1  = (const float*)d_in[3];
    const float* al1 = (const float*)d_in[4];
    const float* ar1 = (const float*)d_in[5];
    const float* b1  = (const float*)d_in[6];
    const float* W2  = (const float*)d_in[7];
    const float* al2 = (const float*)d_in[8];
    const float* ar2 = (const float*)d_in[9];
    const float* b2  = (const float*)d_in[10];
    const float* lw[4] = {(const float*)d_in[11], (const float*)d_in[13],
                          (const float*)d_in[15], (const float*)d_in[17]};
    const float* lb[4] = {(const float*)d_in[12], (const float*)d_in[14],
                          (const float*)d_in[16], (const float*)d_in[18]};
    const float* lw5 = (const float*)d_in[19];
    const float* lb5 = (const float*)d_in[20];
    float* out = (float*)d_out;

    char* ws = (char*)d_ws;
    size_t off = 0;
    auto alloc = [&](size_t bytes) { size_t o = off; off = (off + bytes + 255) & ~(size_t)255; return o; };
    float*    t0    = (float*)(ws + alloc((size_t)NN * HID3 * 4));   // h features (fp32: edge gather)
    float*    agg   = (float*)(ws + alloc((size_t)NN * HID3 * 4));   // aggregation acc
    __bf16*   xb0   = (__bf16*)(ws + alloc((size_t)NN * FDIM * 2));  // bf16 activation ping
    __bf16*   xb1   = (__bf16*)(ws + alloc((size_t)NN * FDIM * 2));  // bf16 activation pong
    float*    el    = (float*)(ws + alloc((size_t)NN * HH * 4));
    float*    er    = (float*)(ws + alloc((size_t)NN * HH * 4));
    unsigned* menc  = (unsigned*)(ws + alloc((size_t)NN * HH * 4));
    float*    den   = (float*)(ws + alloc((size_t)NN * HH * 4));
    float*    esc   = (float*)(ws + alloc((size_t)EE * HH * 4));
    __bf16*   W1T   = (__bf16*)(ws + alloc((size_t)HID3 * FDIM * 2));
    __bf16*   W2T   = (__bf16*)(ws + alloc((size_t)HID3 * FDIM * 2));
    __bf16*   lwT[4];
    for (int i = 0; i < 4; ++i) lwT[i] = (__bf16*)(ws + alloc((size_t)FDIM * FDIM * 2));
    __bf16*   lw5T  = (__bf16*)(ws + alloc((size_t)16 * FDIM * 2));

    auto cdiv = [](int a, int b) { return (a + b - 1) / b; };

    // --- weight prep (fp32 -> transposed bf16) ---
    transpose_pad_bf16<<<cdiv(HID3 * FDIM, 256), 256, 0, stream>>>(W1, W1T, FDIM, HID3, HID3);
    transpose_pad_bf16<<<cdiv(HID3 * FDIM, 256), 256, 0, stream>>>(W2, W2T, FDIM, HID3, HID3);
    for (int i = 0; i < 4; ++i)
        transpose_pad_bf16<<<cdiv(FDIM * FDIM, 256), 256, 0, stream>>>(lw[i], lwT[i], FDIM, FDIM, FDIM);
    transpose_pad_bf16<<<cdiv(16 * FDIM, 256), 256, 0, stream>>>(lw5, lw5T, FDIM, 6, 16);

    const int nodeTiles = NN / 16;                 // 3125 exact
    unsigned mInit;
    { float f = -3.0e38f; unsigned u; __builtin_memcpy(&u, &f, 4);
      mInit = (u & 0x80000000u) ? ~u : (u | 0x80000000u); }

    auto gat_layer = [&](const __bf16* xin, const __bf16* WT,
                         const float* al, const float* ar, const float* b, __bf16* xoutB) {
        // h = x @ W  (N x 384, fp32 out), FG=4 -> featGroups = 24/4 = 6
        gemm_wmma<4><<<cdiv(nodeTiles * 6 * 32, 256), 256, 0, stream>>>(
            xin, WT, nullptr, t0, nullptr, nodeTiles, 6, HID3, HID3, 0.f, 0);
        // el/er
        attn_dots<<<cdiv(NN * HH * 32, 256), 256, 0, stream>>>(t0, al, ar, el, er);
        // reset accumulators
        fill_u32<<<cdiv(NN * HH, 256), 256, 0, stream>>>(menc, mInit, NN * HH);
        fill_f32<<<cdiv(NN * HH, 256), 256, 0, stream>>>(den, 0.f, NN * HH);
        fill_f32<<<cdiv(NN * HID3, 256), 256, 0, stream>>>(agg, 0.f, NN * HID3);
        // edge softmax + aggregate
        edge_scores<<<cdiv(EE, 256), 256, 0, stream>>>(src, dst, el, er, esc, menc);
        edge_expsum<<<cdiv(EE, 256), 256, 0, stream>>>(dst, esc, menc, den);
        edge_agg<<<cdiv(EE * HH * 32, 256), 256, 0, stream>>>(src, dst, t0, esc, den, agg);
        // +bias -> leaky(0.01) -> head mean -> bf16
        bias_act_mean<<<cdiv(NN * FDIM, 256), 256, 0, stream>>>(agg, b, xoutB);
    };

    // in_feat fp32 -> bf16 once
    cvt_bf16<<<cdiv(NN * FDIM / 4, 256), 256, 0, stream>>>(in_feat, xb0, NN * FDIM / 4);

    gat_layer(xb0, W1T, al1, ar1, b1, xb1);
    gat_layer(xb1, W2T, al2, ar2, b2, xb0);

    // --- MLP head: 4x (128->128, bias, leaky 0.01), bf16 chain; FG=4 -> featGroups = 8/4 = 2 ---
    const int mlpThreads = nodeTiles * 2 * 32;
    gemm_wmma<4><<<cdiv(mlpThreads, 256), 256, 0, stream>>>(xb0, lwT[0], lb[0], nullptr, xb1, nodeTiles, 2, FDIM, FDIM, 0.01f, 1);
    gemm_wmma<4><<<cdiv(mlpThreads, 256), 256, 0, stream>>>(xb1, lwT[1], lb[1], nullptr, xb0, nodeTiles, 2, FDIM, FDIM, 0.01f, 1);
    gemm_wmma<4><<<cdiv(mlpThreads, 256), 256, 0, stream>>>(xb0, lwT[2], lb[2], nullptr, xb1, nodeTiles, 2, FDIM, FDIM, 0.01f, 1);
    gemm_wmma<4><<<cdiv(mlpThreads, 256), 256, 0, stream>>>(xb1, lwT[3], lb[3], nullptr, xb0, nodeTiles, 2, FDIM, FDIM, 0.01f, 1);
    // final 128 -> 6 (padded to one 16-wide tile, store masked to 6 cols), fp32 out, no act
    gemm_wmma<1><<<cdiv(nodeTiles * 1 * 32, 256), 256, 0, stream>>>(
        xb0, lw5T, lb5, out, nullptr, nodeTiles, 1, 6, 6, 0.f, 0);
}
